// PAM_39367670235961
// MI455X (gfx1250) — compile-verified, matrix operand order checked by reading, not run on hardware
//
#include <hip/hip_runtime.h>
#include <hip/hip_bf16.h>

typedef _Float16 half_t;
typedef __attribute__((ext_vector_type(16))) _Float16 v16h;
typedef __attribute__((ext_vector_type(8)))  _Float16 v8h;
typedef __attribute__((ext_vector_type(8)))  float    v8f;
typedef int i32x4 __attribute__((vector_size(16)));

#define B_   4
#define CIN  256
#define CQK  128
#define COUT 256
#define N_   4096   // 64*64
#define STEPS (N_ / 32)

static __device__ __forceinline__ v16h cat16(v8h lo, v8h hi) {
    return __builtin_shufflevector(lo, hi, 0,1,2,3,4,5,6,7,8,9,10,11,12,13,14,15);
}

static __device__ __forceinline__ v8f wmma_f16(v16h a, v16h b, v8f c) {
    return __builtin_amdgcn_wmma_f32_16x16x32_f16(false, a, false, b, (short)0, c, false, false);
}

// CDNA5 async memory->LDS copy (16B), tracked by ASYNCcnt.  Falls back to a
// synchronous VGPR round-trip if the builtin is not declared by this clang.
static __device__ __forceinline__ void async_copy16(const void* gsrc, void* ldst) {
#if __has_builtin(__builtin_amdgcn_global_load_async_to_lds_b128)
    __builtin_amdgcn_global_load_async_to_lds_b128(
        (__attribute__((address_space(1))) i32x4*)gsrc,
        (__attribute__((address_space(3))) i32x4*)ldst, 0, 0);
#else
    *(uint4*)ldst = *(const uint4*)gsrc;
#endif
}

static __device__ __forceinline__ void wait_async0() {
#if __has_builtin(__builtin_amdgcn_s_wait_asynccnt)
    __builtin_amdgcn_s_wait_asynccnt(0);
#endif
}

// ---------------------------------------------------------------------------
// Stage 1: fused QKV projection.  out[n,o] = sum_c x[b,c,n] * W[o,c] + bias[o]
// One wave computes a 16(n) x 16(o) tile with 8 WMMA accumulation steps (Cin=256).
// Q,K stored [b][n][c] (row-major per query/key, f16); V stored [b][cout][n].
// ---------------------------------------------------------------------------
__global__ __launch_bounds__(32) void proj_kernel(
    const float* __restrict__ x,
    const float* __restrict__ Wq, const float* __restrict__ bq,
    const float* __restrict__ Wk, const float* __restrict__ bk,
    const float* __restrict__ Wv, const float* __restrict__ bv,
    half_t* __restrict__ Qh, half_t* __restrict__ Kh, half_t* __restrict__ Vh)
{
    const int lane   = threadIdx.x & 31;
    const int nbase  = blockIdx.x * 16;
    const int obase  = blockIdx.y * 16;     // 0..511 channels: [Q 0-127][K 128-255][V 256-511]
    const int b      = blockIdx.z;

    const float* Wsel; const float* bsel; int oloc;
    if (obase < 128)      { Wsel = Wq; bsel = bq; oloc = obase; }
    else if (obase < 256) { Wsel = Wk; bsel = bk; oloc = obase - 128; }
    else                  { Wsel = Wv; bsel = bv; oloc = obase - 256; }

    const int col    = lane & 15;
    const int half16 = lane >> 4;       // which 16-lane half
    const int koff   = half16 ? 8 : 0;  // A/B 16-bit fragment K offset per ISA layout
    const int nrow   = lane & 15;

    v8f acc = {};
    #pragma unroll
    for (int kc = 0; kc < 8; ++kc) {     // Cin = 8 * 32
        const int cbase = kc * 32 + koff;
        v16h a, w;
        #pragma unroll
        for (int j = 0; j < 8; ++j) {
            const int c0 = cbase + j;
            const int c1 = cbase + 16 + j;
            a[j]   = (_Float16)x[((size_t)b * CIN + c0) * N_ + nbase + nrow];
            a[j+8] = (_Float16)x[((size_t)b * CIN + c1) * N_ + nbase + nrow];
            w[j]   = (_Float16)Wsel[(size_t)(oloc + col) * CIN + c0];
            w[j+8] = (_Float16)Wsel[(size_t)(oloc + col) * CIN + c1];
        }
        acc = wmma_f16(a, w, acc);
    }

    const float bias = bsel[oloc + col];
    #pragma unroll
    for (int i = 0; i < 8; ++i) {
        const int m = i + half16 * 8;              // C/D layout: M = i + 8*(lane>=16)
        const _Float16 v = (_Float16)(acc[i] + bias);
        if (obase < 128)
            Qh[((size_t)b * N_ + nbase + m) * CQK + oloc + col] = v;
        else if (obase < 256)
            Kh[((size_t)b * N_ + nbase + m) * CQK + oloc + col] = v;
        else
            Vh[((size_t)b * COUT + oloc + col) * N_ + nbase + m] = v;
    }
}

// ---------------------------------------------------------------------------
// Stage 2: flash attention.  4 waves per block, 16 queries per wave, 64/block.
// 128 steps of 32 keys; double-buffered async (memory->LDS) K/V tile fills;
// online softmax.  Per step: 8 WMMA for S, 16 WMMA for P.V, 1 barrier.
// ---------------------------------------------------------------------------
__global__ __launch_bounds__(128) void attn_kernel(
    const half_t* __restrict__ Qh, const half_t* __restrict__ Kh,
    const half_t* __restrict__ Vh, float* __restrict__ merged)
{
    __shared__ __align__(16) _Float16 Ks[2][32][CQK];   // [buf][key][c]    16 KB
    __shared__ __align__(16) _Float16 Vt[2][COUT][32];  // [buf][cout][key] 32 KB
    __shared__ __align__(16) _Float16 Ps[4][16][32];    // per-wave P        4 KB

    const int tid    = threadIdx.x;
    const int wave   = tid >> 5;
    const int lane   = tid & 31;
    const int b      = blockIdx.y;
    const int qbase  = blockIdx.x * 64 + wave * 16;

    const int col    = lane & 15;
    const int half16 = lane >> 4;
    // A/B 16-bit fragment = two contiguous 8xf16 (16B) pieces at koff, koff+16
    // (koff = 0 for lanes 0-15, 8 for lanes 16-31)

    // Issue async fills of K/V tile for step s into buffer bb (12 x 16B / thread).
    auto issue_tile = [&](int bb, int s) {
        const half_t* ksrc = Kh + ((size_t)b * N_ + s * 32) * CQK;   // contiguous 8KB
        _Float16* kdst = &Ks[bb][0][0];
        #pragma unroll
        for (int i = 0; i < 4; ++i) {
            const int f = tid + i * 128;                 // 512 x 16B
            async_copy16(ksrc + f * 8, kdst + f * 8);
        }
        _Float16* vdst = &Vt[bb][0][0];
        #pragma unroll
        for (int i = 0; i < 8; ++i) {
            const int f = tid + i * 128;                 // 1024 x 16B
            const int row = f >> 2, cc = f & 3;          // V global is [cout][n]
            async_copy16(Vh + ((size_t)b * COUT + row) * N_ + s * 32 + cc * 8,
                         vdst + f * 8);
        }
    };

    // Load Q fragments once (A layout, row = query)
    v16h qa[4];
    {
        const v8h* qrow = (const v8h*)(Qh + ((size_t)b * N_ + qbase + col) * CQK);
        #pragma unroll
        for (int kc = 0; kc < 4; ++kc)
            qa[kc] = cat16(qrow[kc * 4 + half16], qrow[kc * 4 + half16 + 2]);
    }

    v8f o_acc[16];
    #pragma unroll
    for (int t = 0; t < 16; ++t) o_acc[t] = (v8f){};
    float rowmax[8], rowsum[8];
    #pragma unroll
    for (int i = 0; i < 8; ++i) { rowmax[i] = -1e30f; rowsum[i] = 0.0f; }

    issue_tile(0, 0);

    for (int s = 0; s < STEPS; ++s) {
        const int cur = s & 1;
        // My async fills of `cur` (issued last iteration) must be complete,
        // then barrier: everyone's fills of `cur` done AND everyone finished
        // reading the other buffer -> safe to start overwriting it.
        wait_async0();
        __syncthreads();
        if (s + 1 < STEPS) issue_tile(cur ^ 1, s + 1);   // overlaps compute below

        // --- S = Q . K^T for two 16-key subtiles ---
        v8f s0 = (v8f){}, s1 = (v8f){};
        const v8h* kr0 = (const v8h*)&Ks[cur][col][0];
        const v8h* kr1 = (const v8h*)&Ks[cur][col + 16][0];
        #pragma unroll
        for (int kc = 0; kc < 4; ++kc) {
            v16h b0 = cat16(kr0[kc * 4 + half16], kr0[kc * 4 + half16 + 2]);
            v16h b1 = cat16(kr1[kc * 4 + half16], kr1[kc * 4 + half16 + 2]);
            s0 = wmma_f16(qa[kc], b0, s0);
            s1 = wmma_f16(qa[kc], b1, s1);
        }

        // --- online softmax (rows spread across 16-lane groups) ---
        float p0[8], p1[8];
        #pragma unroll
        for (int i = 0; i < 8; ++i) {
            float m = fmaxf(s0[i], s1[i]);
            #pragma unroll
            for (int msk = 8; msk >= 1; msk >>= 1)
                m = fmaxf(m, __shfl_xor(m, msk, 32));
            const float nm    = fmaxf(rowmax[i], m);
            const float scale = __expf(rowmax[i] - nm);
            const float e0 = __expf(s0[i] - nm);
            const float e1 = __expf(s1[i] - nm);
            float rs = e0 + e1;
            #pragma unroll
            for (int msk = 8; msk >= 1; msk >>= 1)
                rs += __shfl_xor(rs, msk, 32);
            rowsum[i] = rowsum[i] * scale + rs;
            rowmax[i] = nm;
            p0[i] = e0; p1[i] = e1;
            #pragma unroll
            for (int t = 0; t < 16; ++t) o_acc[t][i] *= scale;
        }

        // --- transpose P (D layout -> A layout) through per-wave LDS ---
        // Ps is private to this wave, and LDS ops from one wave complete in
        // order (DScnt); only compiler reordering must be fenced.
        #pragma unroll
        for (int i = 0; i < 8; ++i) {
            const int m = i + half16 * 8;
            Ps[wave][m][col]      = (_Float16)p0[i];
            Ps[wave][m][col + 16] = (_Float16)p1[i];
        }
        asm volatile("" ::: "memory");

        // --- O += P . V ---
        const v8h* pr = (const v8h*)&Ps[wave][col][0];
        const v16h pa = cat16(pr[half16], pr[half16 + 2]);
        #pragma unroll
        for (int ct = 0; ct < 16; ++ct) {
            const v8h* vr = (const v8h*)&Vt[cur][ct * 16 + col][0];
            const v16h vb = cat16(vr[half16], vr[half16 + 2]);
            o_acc[ct] = wmma_f16(pa, vb, o_acc[ct]);
        }
    }

    // --- normalize and write merged [b][cout][n] f32 ---
    #pragma unroll
    for (int ct = 0; ct < 16; ++ct) {
        #pragma unroll
        for (int i = 0; i < 8; ++i) {
            const int m = i + half16 * 8;
            merged[((size_t)b * COUT + ct * 16 + col) * N_ + qbase + m] =
                o_acc[ct][i] / rowsum[i];
        }
    }
}

// ---------------------------------------------------------------------------
// Stage 3: training-mode BatchNorm over (B,H,W) per channel, biased variance.
// ---------------------------------------------------------------------------
__global__ __launch_bounds__(256) void bn_kernel(
    const float* __restrict__ merged, const float* __restrict__ gamma,
    const float* __restrict__ beta, float* __restrict__ out)
{
    __shared__ float s1s[256], s2s[256];
    const int c = blockIdx.x, t = threadIdx.x;
    float s1 = 0.0f, s2 = 0.0f;
    for (int b = 0; b < B_; ++b) {
        const float* p = merged + ((size_t)b * COUT + c) * N_;
        for (int n = t; n < N_; n += 256) {
            const float v = p[n];
            s1 += v; s2 += v * v;
        }
    }
    s1s[t] = s1; s2s[t] = s2;
    __syncthreads();
    for (int off = 128; off > 0; off >>= 1) {
        if (t < off) { s1s[t] += s1s[t + off]; s2s[t] += s2s[t + off]; }
        __syncthreads();
    }
    const float inv_n = 1.0f / (float)(B_ * N_);
    const float mean  = s1s[0] * inv_n;
    const float var   = s2s[0] * inv_n - mean * mean;
    const float istd  = rsqrtf(var + 1e-5f);
    const float gg    = gamma[c] * istd;
    const float bb    = beta[c] - mean * gg;
    for (int b = 0; b < B_; ++b) {
        const float* p = merged + ((size_t)b * COUT + c) * N_;
        float* o       = out    + ((size_t)b * COUT + c) * N_;
        for (int n = t; n < N_; n += 256)
            o[n] = p[n] * gg + bb;
    }
}

extern "C" void kernel_launch(void* const* d_in, const int* in_sizes, int n_in,
                              void* d_out, int out_size, void* d_ws, size_t ws_size,
                              hipStream_t stream) {
    const float* x     = (const float*)d_in[0];
    const float* Wq    = (const float*)d_in[1];
    const float* bq    = (const float*)d_in[2];
    const float* Wk    = (const float*)d_in[3];
    const float* bk    = (const float*)d_in[4];
    const float* Wv    = (const float*)d_in[5];
    const float* bv    = (const float*)d_in[6];
    const float* gamma = (const float*)d_in[7];
    const float* beta  = (const float*)d_in[8];

    char* ws = (char*)d_ws;
    half_t* Qh     = (half_t*)(ws);                        //  4 MB
    half_t* Kh     = (half_t*)(ws + ((size_t)4  << 20));   //  4 MB
    half_t* Vh     = (half_t*)(ws + ((size_t)8  << 20));   //  8 MB
    float*  merged = (float*) (ws + ((size_t)16 << 20));   // 16 MB

    proj_kernel<<<dim3(N_ / 16, (CQK + CQK + COUT) / 16, B_), 32, 0, stream>>>(
        x, Wq, bq, Wk, bk, Wv, bv, Qh, Kh, Vh);
    attn_kernel<<<dim3(N_ / 64, B_), 128, 0, stream>>>(Qh, Kh, Vh, merged);
    bn_kernel<<<dim3(COUT), 256, 0, stream>>>(merged, gamma, beta, (float*)d_out);
}
